// Attention_25778393710717
// MI455X (gfx1250) — matrix operand order repaired and checked by reading, not compiled, not run on hardware
//
#include <hip/hip_runtime.h>

#define NSEQ   2048
#define DIMX   128
#define HEADS_ 32
#define DHEAD  256          // INNER / HEADS
#define INNER_ 8192
#define CCOLS  24576        // 3 * INNER
#define QKV_BYTES (33554432ull)  // HEADS*NSEQ*DHEAD*2 = 32 MB

typedef __attribute__((ext_vector_type(16))) _Float16 v16h;
typedef __attribute__((ext_vector_type(8)))  _Float16 v8h;
typedef __attribute__((ext_vector_type(8)))  float    v8f;
typedef int v4i __attribute__((__vector_size__(16)));   // matches builtin param type

#define AS1 __attribute__((address_space(1)))
#define AS3 __attribute__((address_space(3)))

#if defined(__AMDGCN__) && __has_builtin(__builtin_amdgcn_global_load_async_to_lds_b128)
#define HAVE_ASYNC_LDS 1
#else
#define HAVE_ASYNC_LDS 0
#endif

__device__ __forceinline__ void wait_async_zero() {
#if defined(__AMDGCN__)
#if __has_builtin(__builtin_amdgcn_s_wait_asynccnt)
  __builtin_amdgcn_s_wait_asynccnt(0);
#else
  asm volatile("s_wait_asynccnt 0" ::: "memory");
#endif
#endif
}

__device__ __forceinline__ v8f vzero8() {
  v8f z;
#pragma unroll
  for (int i = 0; i < 8; ++i) z[i] = 0.f;
  return z;
}

__device__ __forceinline__ v8f wmma_f32_f16(v16h a, v16h b, v8f c) {
  return __builtin_amdgcn_wmma_f32_16x16x32_f16(false, a, false, b, (short)0, c,
                                                false, false);
}

// 16 contiguous halves (B fragment per-lane payload)
__device__ __forceinline__ v16h load16h(const _Float16* p) {
  v8h lo = *(const v8h*)p;
  v8h hi = *(const v8h*)(p + 8);
  v16h r;
#pragma unroll
  for (int i = 0; i < 8; ++i) { r[i] = lo[i]; r[i + 8] = hi[i]; }
  return r;
}

// A fragment per-lane payload: 16 contiguous halves at p
__device__ __forceinline__ v16h load_a16(const _Float16* p) {
  v8h lo = *(const v8h*)p;
  v8h hi = *(const v8h*)(p + 8);
  v16h r;
#pragma unroll
  for (int i = 0; i < 8; ++i) { r[i] = lo[i]; r[i + 8] = hi[i]; }
  return r;
}

// A 16x32 fragment from row-major f32 matrix, converting to f16
__device__ __forceinline__ v16h load_a_f32c(const float* base, int row0, int ld,
                                            int k0, int lane) {
  int r  = row0 + (lane & 15);
  int kk = k0 + ((lane >> 4) << 4);
  const float* p = base + (size_t)r * ld + kk;
  v16h a;
#pragma unroll
  for (int e = 0; e < 8; ++e) a[e] = (_Float16)p[e];
#pragma unroll
  for (int e = 0; e < 8; ++e) a[e + 8] = (_Float16)p[e + 8];
  return a;
}

// B 32x16 fragment where B(kk,n) = src[col0+n][kk] (row-major f32 source, i.e. B = srcT)
__device__ __forceinline__ v16h load_bT_f32(const float* base, int col0, int ld,
                                            int k0, int lane) {
  int c  = col0 + (lane & 15);
  int kk = k0 + ((lane >> 4) << 4);
  const float* p = base + (size_t)c * ld + kk;
  v16h b;
#pragma unroll
  for (int e = 0; e < 16; ++e) b[e] = (_Float16)p[e];
  return b;
}

// ---------------- Kernel 1: QKV GEMM (x @ Wqkv^T) -> f16 q/k/v [h][n][d] --------
__global__ __launch_bounds__(128) void qkv_kernel(
    const float* __restrict__ x, const float* __restrict__ Wqkv,
    _Float16* __restrict__ qf, _Float16* __restrict__ kf,
    _Float16* __restrict__ vf) {
  const int lane = threadIdx.x & 31;
  const int wave = threadIdx.x >> 5;
  const int c0 = blockIdx.x * 64 + wave * 16;
  const int n0 = blockIdx.y * 16;

  v8f acc = vzero8();
#pragma unroll
  for (int k0 = 0; k0 < DIMX; k0 += 32) {
    v16h a = load_a_f32c(x, n0, DIMX, k0, lane);
    v16h b = load_bT_f32(Wqkv, c0, DIMX, k0, lane);
    acc = wmma_f32_f16(a, b, acc);
  }

  const int c   = c0 + (lane & 15);
  const int sel = c >> 13;       // 0=q 1=k 2=v
  const int rem = c & 8191;
  const int h   = rem >> 8;
  const int d   = rem & 255;
  _Float16* dst = (sel == 0) ? qf : ((sel == 1) ? kf : vf);
  _Float16* p = dst + (size_t)h * NSEQ * DHEAD + d;
  const int rb = (lane >> 4) << 3;
#pragma unroll
  for (int r = 0; r < 8; ++r) {
    int row = n0 + rb + r;
    p[(size_t)row * DHEAD] = (_Float16)acc[r];
  }
}

// ---------------- Kernel 2: rotary on first 64 dims + fold SCALE into q --------
__global__ void rotary_kernel(const float* __restrict__ rope,
                              _Float16* __restrict__ qf,
                              _Float16* __restrict__ kf) {
  const int row = blockIdx.x;           // h*NSEQ + n
  const int n   = row & (NSEQ - 1);
  const int j   = threadIdx.x;          // 0..223
  const size_t base = (size_t)row * DHEAD;
  const float scale = 0.08838834764831845f;  // 128^-0.5

  if (j < 32) {
    float f0 = rope[n * 64 + j];
    float f1 = rope[n * 64 + j + 32];
    float c0 = __cosf(f0), s0 = __sinf(f0);
    float c1 = __cosf(f1), s1 = __sinf(f1);
    float q0 = (float)qf[base + j], q1 = (float)qf[base + j + 32];
    float k0 = (float)kf[base + j], k1 = (float)kf[base + j + 32];
    qf[base + j]      = (_Float16)((q0 * c0 - q1 * s0) * scale);
    qf[base + j + 32] = (_Float16)((q1 * c1 + q0 * s1) * scale);
    kf[base + j]      = (_Float16)(k0 * c0 - k1 * s0);
    kf[base + j + 32] = (_Float16)(k1 * c1 + k0 * s1);
  } else {
    int d = j + 32;                     // 64..255
    qf[base + d] = (_Float16)((float)qf[base + d] * scale);
  }
}

// ---------------- Kernel 3: flash attention per head ---------------------------
// grid: (NSEQ/16, HEADS), block 128 (4 waves). Wave w owns d-range [w*64, w*64+64).
__global__ __launch_bounds__(128) void attn_kernel(
    const _Float16* __restrict__ qf, const _Float16* __restrict__ kf,
    const _Float16* __restrict__ vf, _Float16* __restrict__ of) {
  const int lane = threadIdx.x & 31;
  const int wave = threadIdx.x >> 5;
  const int q0 = blockIdx.x * 16;
  const int h  = blockIdx.y;

  __shared__ __align__(16) _Float16 sK[32][DHEAD];    // 16 KB
  __shared__ __align__(16) _Float16 sVt[DHEAD][32];   // 16 KB (transposed V)
  __shared__ __align__(16) float    sS[4][16][32];    // 8 KB partial S
  __shared__ __align__(16) _Float16 sP[16][32];       // 1 KB
  __shared__ float sM[16], sL[16], sScale[16];

  const _Float16* qh = qf + (size_t)h * NSEQ * DHEAD;
  const _Float16* kh = kf + (size_t)h * NSEQ * DHEAD;
  const _Float16* vh = vf + (size_t)h * NSEQ * DHEAD;

  // Q fragments for this wave's 64-dim slice (rows q0..q0+15)
  v16h qa[2];
#pragma unroll
  for (int cc = 0; cc < 2; ++cc) {
    const _Float16* p = qh + (size_t)(q0 + (lane & 15)) * DHEAD + wave * 64 +
                        cc * 32 + ((lane >> 4) << 4);
    qa[cc] = load_a16(p);
  }

  v8f o[4];
#pragma unroll
  for (int dt = 0; dt < 4; ++dt) o[dt] = vzero8();

  if (threadIdx.x < 16) { sM[threadIdx.x] = -1e30f; sL[threadIdx.x] = 0.f; }

  const int rb = (lane >> 4) << 3;

  for (int j0 = 0; j0 < NSEQ; j0 += 32) {
    __syncthreads();  // previous iteration fully consumed sK/sVt/sP; sM/sL init visible

    // ---- stage K tile (async mem->LDS) and V tile (transposed) into LDS ----
    for (int idx = threadIdx.x; idx < 32 * (DHEAD / 8); idx += 128) {
      int key = idx >> 5;               // 0..31
      int off = (idx & 31) << 3;        // 0..248 step 8 halves (16B)
      const _Float16* gk = kh + (size_t)(j0 + key) * DHEAD + off;
#if HAVE_ASYNC_LDS
      __builtin_amdgcn_global_load_async_to_lds_b128(
          (AS1 v4i*)gk, (AS3 v4i*)&sK[key][off], 0, 0);
#else
      *(v8h*)&sK[key][off] = *(const v8h*)gk;
#endif
      v8h tv = *(const v8h*)(vh + (size_t)(j0 + key) * DHEAD + off);
#pragma unroll
      for (int e = 0; e < 8; ++e) sVt[off + e][key] = tv[e];
    }
    if (j0 + 32 < NSEQ)  // hint next K tile toward L2
      __builtin_prefetch(kh + (size_t)(j0 + 32 + (threadIdx.x & 31)) * DHEAD, 0, 1);
#if HAVE_ASYNC_LDS
    wait_async_zero();
#endif
    __syncthreads();

    // ---- partial S = Q_slice * K_slice^T (this wave's 64 dims, 32 keys) ----
    v8f s0 = vzero8(), s1 = vzero8();
#pragma unroll
    for (int cc = 0; cc < 2; ++cc) {
      int d0 = wave * 64 + cc * 32 + ((lane >> 4) << 4);
      v16h bk0 = load16h(&sK[(lane & 15)][d0]);
      v16h bk1 = load16h(&sK[16 + (lane & 15)][d0]);
      s0 = wmma_f32_f16(qa[cc], bk0, s0);
      s1 = wmma_f32_f16(qa[cc], bk1, s1);
    }
    const int colb = lane & 15;
#pragma unroll
    for (int r = 0; r < 8; ++r) {
      sS[wave][rb + r][colb]      = s0[r];
      sS[wave][rb + r][16 + colb] = s1[r];
    }
    __syncthreads();

    // ---- reduce partials across waves + online softmax stats (wave 0) ----
    if (threadIdx.x < 32) {
      int r  = threadIdx.x & 15;       // row
      int hh = threadIdx.x >> 4;       // half: cols [hh*16, hh*16+16)
      float mold = sM[r];
      float mrow = mold;
      float sv[16];
#pragma unroll
      for (int c2 = 0; c2 < 16; ++c2) {
        int col = hh * 16 + c2;
        float s = sS[0][r][col] + sS[1][r][col] + sS[2][r][col] + sS[3][r][col];
        sv[c2] = s;
        mrow = fmaxf(mrow, s);
      }
      mrow = fmaxf(mrow, __shfl_xor(mrow, 16, 32));  // combine halves (wave32)
      float lsum = 0.f;
#pragma unroll
      for (int c2 = 0; c2 < 16; ++c2) {
        float pv = __expf(sv[c2] - mrow);
        sP[r][hh * 16 + c2] = (_Float16)pv;
        lsum += pv;
      }
      lsum += __shfl_xor(lsum, 16, 32);
      if (hh == 0) {
        float scl = __expf(mold - mrow);
        sM[r] = mrow;
        sScale[r] = scl;
        sL[r] = sL[r] * scl + lsum;
      }
    }
    __syncthreads();

    // ---- rescale O and accumulate P * V_slice ----
    float scl[8];
#pragma unroll
    for (int r = 0; r < 8; ++r) scl[r] = sScale[rb + r];
    v16h pa = load_a16(&sP[lane & 15][(lane >> 4) << 4]);
#pragma unroll
    for (int dt = 0; dt < 4; ++dt) {
      int d = wave * 64 + dt * 16 + (lane & 15);
      v16h bv = load16h(&sVt[d][(lane >> 4) << 4]);
#pragma unroll
      for (int r = 0; r < 8; ++r) o[dt][r] *= scl[r];
      o[dt] = wmma_f32_f16(pa, bv, o[dt]);
    }
  }

  // ---- epilogue: normalize by l and store f16 ----
  float invl[8];
#pragma unroll
  for (int r = 0; r < 8; ++r) invl[r] = 1.f / sL[rb + r];
  _Float16* oh = of + (size_t)h * NSEQ * DHEAD;
#pragma unroll
  for (int dt = 0; dt < 4; ++dt) {
    int d = wave * 64 + dt * 16 + (lane & 15);
#pragma unroll
    for (int r = 0; r < 8; ++r)
      oh[(size_t)(q0 + rb + r) * DHEAD + d] = (_Float16)(o[dt][r] * invl[r]);
  }
}

// ---------------- Kernel 4: projection (attn_out @ Wproj^T + b) ----------------
// grid: (DIMX/16, NSEQ/64), block 128; wave owns one 16-row n tile.
__global__ __launch_bounds__(128) void proj_kernel(
    const _Float16* __restrict__ of, const float* __restrict__ Wproj,
    const float* __restrict__ bproj, float* __restrict__ out) {
  const int lane = threadIdx.x & 31;
  const int wave = threadIdx.x >> 5;
  const int c0 = blockIdx.x * 16;
  const int n0 = (blockIdx.y * 4 + wave) * 16;

  v8f acc = vzero8();
  for (int k0 = 0; k0 < INNER_; k0 += 32) {
    // A: attn_out rows; flat i = h*256 + d maps to [h][n][d] layout
    int r  = n0 + (lane & 15);
    int kk = k0 + ((lane >> 4) << 4);
    int h = kk >> 8, d = kk & 255;
    v16h a = load_a16(of + ((size_t)h * NSEQ + r) * DHEAD + d);
    v16h b = load_bT_f32(Wproj, c0, INNER_, k0, lane);
    acc = wmma_f32_f16(a, b, acc);
  }

  const int c = c0 + (lane & 15);
  const float bias = bproj[c];
  const int rb = (lane >> 4) << 3;
#pragma unroll
  for (int r = 0; r < 8; ++r)
    out[(size_t)(n0 + rb + r) * DIMX + c] = acc[r] + bias;
}

// -------------------------------------------------------------------------------
extern "C" void kernel_launch(void* const* d_in, const int* in_sizes, int n_in,
                              void* d_out, int out_size, void* d_ws, size_t ws_size,
                              hipStream_t stream) {
  const float* x     = (const float*)d_in[0];
  const float* Wqkv  = (const float*)d_in[1];
  const float* Wproj = (const float*)d_in[2];
  const float* bproj = (const float*)d_in[3];
  const float* rope  = (const float*)d_in[4];
  float* out = (float*)d_out;

  char* ws = (char*)d_ws;
  _Float16* qf = (_Float16*)(ws);
  _Float16* kf = (_Float16*)(ws + QKV_BYTES);
  _Float16* vf = (_Float16*)(ws + 2 * QKV_BYTES);
  _Float16* of = (_Float16*)(ws + 3 * QKV_BYTES);

  qkv_kernel<<<dim3(CCOLS / 64, NSEQ / 16), 128, 0, stream>>>(x, Wqkv, qf, kf, vf);
  rotary_kernel<<<dim3(HEADS_ * NSEQ), 224, 0, stream>>>(rope, qf, kf);
  attn_kernel<<<dim3(NSEQ / 16, HEADS_), 128, 0, stream>>>(qf, kf, vf, of);
  proj_kernel<<<dim3(DIMX / 16, NSEQ / 64), 128, 0, stream>>>(of, Wproj, bproj, out);
}
